// TernaryConv2d_36369783063104
// MI455X (gfx1250) — matrix operand order repaired and checked by reading, not compiled, not run on hardware
//
#include <hip/hip_runtime.h>
#include <hip/hip_fp16.h>

typedef __attribute__((ext_vector_type(16))) _Float16 v16h;
typedef __attribute__((ext_vector_type(8)))  _Float16 v8h;
typedef __attribute__((ext_vector_type(8)))  float    v8f;

#define B_     32
#define CIN_   128
#define COUT_  256
#define H_     56
#define W_     56
#define HW_    (H_ * W_)          // 3136
#define NPIX_  (B_ * HW_)         // 100352
#define KDIM_  (CIN_ * 9)         // 1152, packed r-major: K = r*128 + ci
#define NSTEP_ (KDIM_ / 32)       // 36

#define TN 64          // pixels per workgroup (4 subtiles x 16)
#define NWAVE 16       // waves per workgroup -> TM = 256 (all couts)
#define NTHREADS (NWAVE * 32)
#define LDS_STRIDE 48  // halves per pixel row (96B: keeps 16B-aligned frag reads)

#define CAT16(lo, hi) __builtin_shufflevector((lo), (hi), 0, 1, 2, 3, 4, 5, 6, 7, \
                                              8, 9, 10, 11, 12, 13, 14, 15)

// ---------------- weight prep: logits -> E[w], Var[w], f16, r-major K packing ----------------
__global__ __launch_bounds__(256) void ternary_prep_weights(
    const float* __restrict__ logits,   // (COUT, CIN, 3, 3, 2)
    _Float16* __restrict__ wmean,       // (COUT, 9, CIN) : K = r*128 + ci
    _Float16* __restrict__ wvar)
{
    int idx = blockIdx.x * 256 + threadIdx.x;    // output index, r-major K
    if (idx >= COUT_ * KDIM_) return;
    int co = idx / KDIM_;
    int t  = idx - co * KDIM_;
    int r  = t >> 7;                             // kh*3+kw
    int ci = t & 127;
    int lidx = (co * KDIM_ + ci * 9 + r) * 2;    // source (ci-major K) logits
    float ml = logits[lidx + 0];
    float sl = logits[lidx + 1];
    float pm = 1.0f / (1.0f + __expf(-ml));      // P(|w|=1)
    float ps = 1.0f / (1.0f + __expf(-sl));      // P(sign=+1 | |w|=1)
    float mean = 2.0f * pm * ps - pm;            // E[w]
    float d    = 2.0f * ps - 1.0f;
    float var  = pm - pm * pm * d * d;           // Var[w]
    wmean[idx] = (_Float16)mean;
    wvar[idx]  = (_Float16)var;
}

// ---------------- fused dual implicit-GEMM conv + reparam epilogue ----------------
__global__ __launch_bounds__(NTHREADS) void ternary_conv_wmma(
    const float*    __restrict__ x,       // (B, CIN, H, W)
    const _Float16* __restrict__ wmean,   // (COUT, KDIM_) r-major K
    const _Float16* __restrict__ wvar,
    const float*    __restrict__ scale,   // (COUT)
    const float*    __restrict__ shift,   // (COUT)
    const float*    __restrict__ noise,   // (B, COUT, H, W)
    float*          __restrict__ out)     // (B, COUT, H, W)
{
    __shared__ _Float16 xt[2][TN * LDS_STRIDE];   // double-buffered im2col tile (f16)

    const int tid  = threadIdx.x;
    const int lane = tid & 31;
    const int wave = tid >> 5;              // 0..15 -> 16-cout band
    const int pixBase = blockIdx.x * TN;    // 1568 blocks over N; all 256 couts here

    // ---- A-fragment addressing (16-bit A 16x32 layout) ----
    const int arow = lane & 15;
    const int ak   = (lane >> 4) * 8;
    const _Float16* wmRow = wmean + (size_t)(wave * 16 + arow) * KDIM_;
    const _Float16* wvRow = wvar  + (size_t)(wave * 16 + arow) * KDIM_;

    // ---- B-fragment addressing (16-bit B 32x16 layout) ----
    const int bn = lane & 15;
    const int bk = (lane >> 4) * 8;

    // ---- staging metadata ----
    // Each thread stages pixel (wave + 16*i) for i<4 with channel kk = lane.
    // All 32 lanes of a wave stage the SAME pixels -> pixel decode & halo masks
    // are wave-uniform: force them scalar via readfirstlane (SALU + s_cbranch,
    // no EXEC juggling on the predicated loads).
    const int kk    = lane;
    const int swave = __builtin_amdgcn_readfirstlane(wave);
    int pfix  [4];   // vector: scalar pixel base + kk*HW_
    int ldsoff[4];   // vector: LDS half-offset
    int vm    [4];   // scalar: bit r set if (oh+kh-1, ow+kw-1) in-bounds
#pragma unroll
    for (int i = 0; i < 4; ++i) {
        int pix = swave + 16 * i;                 // scalar
        int p   = pixBase + pix;                  // scalar
        int b   = p / HW_;                        // scalar magic-mul
        int rem = p - b * HW_;
        int oh  = rem / W_;
        int ow  = rem - oh * W_;
        int pbase = b * CIN_ * HW_ + oh * W_ + ow;        // scalar
        pfix[i]   = pbase + kk * HW_;                     // 1 VALU mad
        ldsoff[i] = pix * LDS_STRIDE + kk;                // 1 VALU add
        int m = 0;
#pragma unroll
        for (int r = 0; r < 9; ++r) {
            int ih = oh + (r / 3) - 1;
            int iw = ow + (r % 3) - 1;
            if ((unsigned)ih < (unsigned)H_ && (unsigned)iw < (unsigned)W_)
                m |= (1 << r);
        }
        vm[i] = m;                                        // scalar
    }

    // predicated (scalar-branch), scalar-folded global fetch for step s
    auto fetchv = [&](int s, float vr[4]) {
        int r      = s >> 2;
        int cibase = (s & 3) * 32;
        int kh     = r / 3;
        int kw     = r - 3 * kh;
        int soff   = cibase * HW_ + (kh - 1) * W_ + (kw - 1);  // uniform -> SADDR/offset
        int rbit   = 1 << r;
        const float* xs = x + soff;
#pragma unroll
        for (int i = 0; i < 4; ++i) {
            vr[i] = 0.0f;
            if (vm[i] & rbit)                   // wave-uniform -> s_cbranch
                vr[i] = xs[pfix[i]];
        }
    };
    auto stagev = [&](int s, const float vr[4]) {
        _Float16* buf = xt[s & 1];
#pragma unroll
        for (int i = 0; i < 4; ++i)
            buf[ldsoff[i]] = (_Float16)vr[i];
    };

    v8f accM[4] = {};
    v8f accV[4] = {};

    // ---- pipeline prologue ----
    {
        float v0[4];
        fetchv(0, v0);
        stagev(0, v0);
    }
    __syncthreads();

    for (int step = 0; step < NSTEP_; ++step) {
        const int k0 = step * 32;

        // issue next step's global loads early; latency hides under WMMAs
        float vnext[4];
        if (step + 1 < NSTEP_) {
            fetchv(step + 1, vnext);
            __builtin_prefetch(wmRow + k0 + 32 + ak, 0, 1);
            __builtin_prefetch(wvRow + k0 + 32 + ak, 0, 1);
        }

        // ---- A fragments (L2-resident packed weights) ----
        v8h m0 = *(const v8h*)(wmRow + k0 + ak);
        v8h m1 = *(const v8h*)(wmRow + k0 + ak + 16);
        v8h w0 = *(const v8h*)(wvRow + k0 + ak);
        v8h w1 = *(const v8h*)(wvRow + k0 + ak + 16);
        v16h aM = CAT16(m0, m1);
        v16h aV = CAT16(w0, w1);

        // ---- 8 WMMAs per wave per step; x^2 fragment = bM*bM (co-executes) ----
        const _Float16* buf = xt[step & 1];
#pragma unroll
        for (int s = 0; s < 4; ++s) {
            const int rowOff = (s * 16 + bn) * LDS_STRIDE + bk;
            v8h l0 = *(const v8h*)(buf + rowOff);
            v8h l1 = *(const v8h*)(buf + rowOff + 16);
            v16h bM = CAT16(l0, l1);
            v16h bV = bM * bM;                      // v_pk_mul_f16 x8
            accM[s] = __builtin_amdgcn_wmma_f32_16x16x32_f16(
                false, aM, false, bM, (short)0, accM[s], false, false);
            accV[s] = __builtin_amdgcn_wmma_f32_16x16x32_f16(
                false, aV, false, bV, (short)0, accV[s], false, false);
        }

        // ---- stage next tile into the other buffer, then single barrier ----
        if (step + 1 < NSTEP_) stagev(step + 1, vnext);
        __syncthreads();
    }

    // ---- epilogue: out = (mean + sqrt(var)*noise) * scale + shift ----
    // C/D layout: VGPR r, lanes 0-15 -> M=r, lanes 16-31 -> M=r+8; N = lane & 15
    const int co0 = wave * 16 + ((lane >> 4) << 3);
    float sc[8], sh[8];
#pragma unroll
    for (int r = 0; r < 8; ++r) {
        sc[r] = scale[co0 + r];
        sh[r] = shift[co0 + r];
    }
#pragma unroll
    for (int s = 0; s < 4; ++s) {
        int p   = pixBase + s * 16 + (lane & 15);
        int b   = p / HW_;
        int rem = p - b * HW_;
        int oh  = rem / W_;
        int ow  = rem - oh * W_;
        size_t base = ((size_t)b * COUT_ + co0) * HW_ + oh * W_ + ow;
#pragma unroll
        for (int r = 0; r < 8; ++r) {
            size_t oidx = base + (size_t)r * HW_;
            float m  = accM[s][r];
            float vv = accV[s][r];
            float val = m + __builtin_sqrtf(fmaxf(vv, 0.0f)) * noise[oidx];
            out[oidx] = val * sc[r] + sh[r];
        }
    }
}

extern "C" void kernel_launch(void* const* d_in, const int* in_sizes, int n_in,
                              void* d_out, int out_size, void* d_ws, size_t ws_size,
                              hipStream_t stream) {
    const float* x      = (const float*)d_in[0];
    const float* logits = (const float*)d_in[1];
    const float* scale  = (const float*)d_in[2];
    const float* shift  = (const float*)d_in[3];
    const float* noise  = (const float*)d_in[4];
    float* out          = (float*)d_out;

    _Float16* wmean = (_Float16*)d_ws;
    _Float16* wvar  = wmean + (size_t)COUT_ * KDIM_;

    {
        int total  = COUT_ * KDIM_;                          // 294912
        int blocks = (total + 255) / 256;
        ternary_prep_weights<<<blocks, 256, 0, stream>>>(logits, wmean, wvar);
    }
    {
        dim3 grid(NPIX_ / TN, 1, 1);                         // 1568 blocks, TM=256 inside
        ternary_conv_wmma<<<grid, NTHREADS, 0, stream>>>(x, wmean, wvar, scale,
                                                         shift, noise, out);
    }
}
